// Head_6021544149785
// MI455X (gfx1250) — matrix operand order repaired and checked by reading, not compiled, not run on hardware
//
#include <hip/hip_runtime.h>
#include <hip/hip_bf16.h>
#include <stdint.h>

// Attention head forward, fused flash-attention style for MI455X (gfx1250).
// Stage 0: pre-convert Wq/Wk/Wv to f16 in WMMA B-fragment order (one 32B vector
//          load per fragment afterwards).
// Stage 1: qkv projection with v_wmma_f32_16x16x32_f16 (f32 accum), q/k/v -> f16 in d_ws.
// Stage 2: flash attention; K/V tiles staged into LDS by the Tensor Data Mover
//          (tensor_load_to_lds + s_wait_tensorcnt), online softmax, WMMA for
//          S = Q K^T and O += P V. Output fp32.

#define N_EMBD 384
#define HEAD   64
#define BATCH  16
#define SEQ    2048
#define MROWS  (BATCH * SEQ)   // 32768

#define KSTEPS   (N_EMBD / 32)          // 12
#define NTILES   (HEAD / 16)            // 4
#define WTILE_H  512                    // halves per 32x16 B tile in fragment order
#define WFRAG_H  (KSTEPS * NTILES * WTILE_H)  // 24576 halves per matrix

typedef __attribute__((ext_vector_type(16))) _Float16 v16h;
typedef __attribute__((ext_vector_type(8)))  float    v8f;
typedef __attribute__((ext_vector_type(4)))  unsigned int u32x4;
typedef __attribute__((ext_vector_type(8)))  int      i32x8;
typedef __attribute__((ext_vector_type(4)))  int      i32x4;

#define WMMA_F16(A, B, C) \
  __builtin_amdgcn_wmma_f32_16x16x32_f16(false, (A), false, (B), (short)0, (C), false, false)

#if defined(__gfx1250__) && __has_builtin(__builtin_amdgcn_tensor_load_to_lds) && \
    __has_builtin(__builtin_amdgcn_s_wait_tensorcnt)
#define USE_TDM 1
#else
#define USE_TDM 0
#endif

__device__ __forceinline__ v8f v8f_zero() {
  v8f z;
#pragma unroll
  for (int i = 0; i < 8; ++i) z[i] = 0.0f;
  return z;
}

// ---------------- CDNA5 wave32 WMMA fragment gathers (per ISA 7.12.2) -------------
// A 16x32 f16: lanes 0-15 -> M=lane, VGPR j holds K = 2*j (+16 for lanes 16-31); this code's k order, not ISA's

// A fragment from row-major fp32 source (converted to f16).
__device__ __forceinline__ v16h load_a_f32(const float* __restrict__ src,
                                           int row0, int ld, int kbase, int lane) {
  const int hf = lane >> 4, m = lane & 15;
  const float* p = src + (size_t)(row0 + m) * ld + kbase + 16 * hf;
  v16h a;
#pragma unroll
  for (int j = 0; j < 8; ++j) {
    const int k = 2 * j;
    float2 f = *(const float2*)(p + k);
    a[2 * j]     = (_Float16)f.x;
    a[2 * j + 1] = (_Float16)f.y;
  }
  return a;
}

// A fragment from row-major f16 source.
__device__ __forceinline__ v16h load_a_f16(const _Float16* __restrict__ src,
                                           int row0, int ld, int kbase, int lane) {
  const int hf = lane >> 4, m = lane & 15;
  const _Float16* p = src + (size_t)(row0 + m) * ld + kbase + 16 * hf;
  v16h a;
#pragma unroll
  for (int j = 0; j < 8; ++j) {
    const int k = 2 * j;
    a[2 * j]     = p[k];
    a[2 * j + 1] = p[k + 1];
  }
  return a;
}

// ---------------- Stage 0: W -> f16 B-fragment order ------------------------------
// B 32x16 f16, this code's order: lane -> column n = lane&15; VGPR j holds K = 2j (+16 for lanes
// 16-31). Fragment buffer: tile (kstep, nt) at ((kstep*4+nt)*32 + lane)*16 + elem,
// so each lane's whole fragment is 16 contiguous (32B-aligned) halves.

__global__ void __launch_bounds__(256)
prep_w_kernel(const float* __restrict__ Wq, const float* __restrict__ Wk,
              const float* __restrict__ Wv, _Float16* __restrict__ wf) {
  const int tid = blockIdx.x * 256 + threadIdx.x;   // 0 .. 3*12288-1, one K-pair each
  const int mat  = tid / 12288;
  const int rem  = tid % 12288;                     // kstep*1024 + nt*256 + lane*8 + j
  const int kst  = rem >> 10;
  const int nt   = (rem >> 8) & 3;
  const int lane = (rem >> 3) & 31;
  const int j    = rem & 7;
  const float* W = (mat == 0) ? Wq : (mat == 1) ? Wk : Wv;
  const int kdim = kst * 32 + 2 * j + 16 * (lane >> 4);
  const int col  = nt * 16 + (lane & 15);
  const size_t dst = (size_t)mat * WFRAG_H + (size_t)((kst * 4 + nt) * 32 + lane) * 16 + 2 * j;
  wf[dst]     = (_Float16)W[(size_t)kdim * HEAD + col];
  wf[dst + 1] = (_Float16)W[(size_t)(kdim + 1) * HEAD + col];
}

// ----------------------------- Stage 1: QKV projection ----------------------------

__global__ void __launch_bounds__(128)
qkv_kernel(const float* __restrict__ x,
           const _Float16* __restrict__ wf,
           const float* __restrict__ bq, const float* __restrict__ bk,
           const float* __restrict__ bv,
           _Float16* __restrict__ qh, _Float16* __restrict__ kh,
           _Float16* __restrict__ vh) {
  const int lane = threadIdx.x & 31;
  const int wave = threadIdx.x >> 5;
  const int tile = blockIdx.x * 4 + wave;       // 16-row tile of x
  const int row0 = tile * 16;

  const _Float16* wq = wf;
  const _Float16* wk = wf + WFRAG_H;
  const _Float16* wv = wf + 2 * (size_t)WFRAG_H;

  v8f cq[4], ck[4], cv[4];
#pragma unroll
  for (int t = 0; t < 4; ++t) { cq[t] = v8f_zero(); ck[t] = v8f_zero(); cv[t] = v8f_zero(); }

  for (int kst = 0; kst < KSTEPS; ++kst) {
    const v16h a = load_a_f32(x, row0, N_EMBD, kst * 32, lane);
#pragma unroll
    for (int nt = 0; nt < 4; ++nt) {
      const size_t fo = (size_t)((kst * 4 + nt) * 32 + lane) * 16;
      v16h b;
      b = *(const v16h*)&wq[fo];
      cq[nt] = WMMA_F16(a, b, cq[nt]);
      b = *(const v16h*)&wk[fo];
      ck[nt] = WMMA_F16(a, b, ck[nt]);
      b = *(const v16h*)&wv[fo];
      cv[nt] = WMMA_F16(a, b, cv[nt]);
    }
  }

  // C/D layout: VGPR r -> row M = r + 8*(lane>>4), col N = lane&15
  const int hf = lane >> 4, n = lane & 15;
#pragma unroll
  for (int nt = 0; nt < 4; ++nt) {
    const float biq = bq[nt * 16 + n];
    const float bik = bk[nt * 16 + n];
    const float biv = bv[nt * 16 + n];
#pragma unroll
    for (int r = 0; r < 8; ++r) {
      const size_t o = (size_t)(row0 + r + 8 * hf) * HEAD + nt * 16 + n;
      qh[o] = (_Float16)(cq[nt][r] + biq);
      kh[o] = (_Float16)(ck[nt][r] + bik);
      vh[o] = (_Float16)(cv[nt][r] + biv);
    }
  }
}

// ----------------------------- Stage 2: flash attention ---------------------------

#define LDT 72   // LDS row stride (halves) for 32x64 K/V tiles (+16B pad per row)
#define LDP 36   // LDS row stride for per-wave 16x32 P tile

#if USE_TDM
// TDM descriptor (ISA 8.3-8.6): 2D tile 64x32 f16, row stride 64 elements in
// memory, hardware pads +4 DWORDs per 32 DWORDs into LDS -> LDT = 72 halves.
__device__ __forceinline__ void tdm_load_tile(const _Float16* gsrc, uint32_t lds_off) {
  const uint64_t ga = (uint64_t)(uintptr_t)gsrc;
  u32x4 g0;
  g0[0] = 1u;                                  // count=1 (valid), user mode
  g0[1] = lds_off;                             // LDS byte address
  g0[2] = (uint32_t)ga;                        // global_addr[31:0]
  g0[3] = (uint32_t)(ga >> 32) | (2u << 30);   // global_addr[56:32] | type=2
  i32x8 g1;
  g1[0] = (1 << 16)        // data_size = 2 bytes
        | (1 << 20)        // pad_enable
        | (4 << 22)        // pad_interval: 32 DWORDs
        | (3 << 25);       // pad_amount: 4 DWORDs (16B)
  g1[1] = (64 << 16);      // tensor_dim0[15:0] = 64
  g1[2] = (32 << 16);      // tensor_dim1[15:0] = 32
  g1[3] = (64 << 16);      // tile_dim0 = 64
  g1[4] = 32;              // tile_dim1 = 32, tile_dim2 = 0
  g1[5] = 64;              // tensor_dim0_stride[31:0] = 64
  g1[6] = 0;               // stride hi bits / tensor_dim1_stride lo = 0
  g1[7] = 0;
  const i32x4 gz4 = {0, 0, 0, 0};              // groups 2/3: dims 2..4 unused
  const i32x8 gz8 = {0, 0, 0, 0, 0, 0, 0, 0};  // extra group (clang-23 6-arg form)
  __builtin_amdgcn_tensor_load_to_lds(g0, g1, gz4, gz4, gz8, 0);
}
#endif

// B fragment for S = Q K^T : B[k_dim][key n] = Ktile[key][dim]; pairs contiguous in dim.
__device__ __forceinline__ v16h load_b_ktile(const _Float16* t, int keybase,
                                             int dimbase, int lane) {
  const int hf = lane >> 4, n = lane & 15;
  const _Float16* p = t + (keybase + n) * LDT + dimbase + 16 * hf;
  v16h b;
#pragma unroll
  for (int j = 0; j < 8; ++j) { b[2 * j] = p[2 * j]; b[2 * j + 1] = p[2 * j + 1]; }
  return b;
}

// B fragment for O += P V : B[key k][head n] = Vtile[key][n]; strided by LDT.
__device__ __forceinline__ v16h load_b_vtile(const _Float16* t, int n0, int lane) {
  const int hf = lane >> 4, n = lane & 15;
  const _Float16* p = t + (16 * hf) * LDT + n0 + n;
  v16h b;
#pragma unroll
  for (int j = 0; j < 8; ++j) {
    b[2 * j]     = p[(2 * j) * LDT];
    b[2 * j + 1] = p[(2 * j + 1) * LDT];
  }
  return b;
}

// A fragment for P (16x32 f16 tile in per-wave LDS, row-major).
__device__ __forceinline__ v16h load_a_ptile(const _Float16* p0, int lane) {
  const int hf = lane >> 4, m = lane & 15;
  const _Float16* p = p0 + m * LDP + 16 * hf;
  v16h a;
#pragma unroll
  for (int j = 0; j < 8; ++j) {
    const int k = 2 * j;
    a[2 * j]     = p[k];
    a[2 * j + 1] = p[k + 1];
  }
  return a;
}

__global__ void __launch_bounds__(256)
attn_kernel(const _Float16* __restrict__ qh, const _Float16* __restrict__ kh,
            const _Float16* __restrict__ vh, float* __restrict__ out) {
  __shared__ __align__(16) _Float16 Kt[32 * LDT];
  __shared__ __align__(16) _Float16 Vt[32 * LDT];
  __shared__ _Float16 Pt[8 * 16 * LDP];

  const int lane  = threadIdx.x & 31;
  const int wave  = threadIdx.x >> 5;
  const int batch = blockIdx.x >> 4;
  const int qblk  = blockIdx.x & 15;
  const int qrow0 = batch * SEQ + qblk * 128 + wave * 16;  // global row of q tile
  const int kvr0  = batch * SEQ;

  _Float16* pt = &Pt[wave * 16 * LDP];

#if USE_TDM
  // Low 32 bits of a flat LDS pointer are the wave-relative LDS byte offset.
  const uint32_t kt_off = (uint32_t)(uintptr_t)(void*)&Kt[0];
  const uint32_t vt_off = (uint32_t)(uintptr_t)(void*)&Vt[0];
#endif

  // Q fragments for K-dim steps 0..31 and 32..63 (loaded once, reused over all keys)
  const v16h aq0 = load_a_f16(qh, qrow0, HEAD, 0, lane);
  const v16h aq1 = load_a_f16(qh, qrow0, HEAD, 32, lane);

  v8f o0 = v8f_zero(), o1 = v8f_zero(), o2 = v8f_zero(), o3 = v8f_zero();
  float mrow[8], lrow[8];
#pragma unroll
  for (int r = 0; r < 8; ++r) { mrow[r] = -1.0e30f; lrow[r] = 0.0f; }

  const int hf = lane >> 4, n = lane & 15;
  const float scale = 0.125f;  // 64^-0.5

  for (int kb = 0; kb < SEQ; kb += 32) {
    __syncthreads();  // all waves done reading previous K/V tiles
#if USE_TDM
    if (wave == 0) {
      tdm_load_tile(&kh[(size_t)(kvr0 + kb) * HEAD], kt_off);
      tdm_load_tile(&vh[(size_t)(kvr0 + kb) * HEAD], vt_off);
      __builtin_amdgcn_s_wait_tensorcnt(0);
    }
#else
    {
      // cooperative stage: 32x64 halves = 256 x 16B per tile, one uint4 each per thread
      const int e = threadIdx.x;
      const int row = e >> 3, c8 = (e & 7) * 8;
      *(uint4*)&Kt[row * LDT + c8] =
          *(const uint4*)&kh[(size_t)(kvr0 + kb + row) * HEAD + c8];
      *(uint4*)&Vt[row * LDT + c8] =
          *(const uint4*)&vh[(size_t)(kvr0 + kb + row) * HEAD + c8];
    }
#endif
    __syncthreads();

    // S tiles: keys [kb, kb+16) and [kb+16, kb+32), contracting over HEAD=64
    v8f s0 = v8f_zero(), s1 = v8f_zero();
    {
      v16h bf;
      bf = load_b_ktile(Kt, 0, 0, lane);   s0 = WMMA_F16(aq0, bf, s0);
      bf = load_b_ktile(Kt, 0, 32, lane);  s0 = WMMA_F16(aq1, bf, s0);
      bf = load_b_ktile(Kt, 16, 0, lane);  s1 = WMMA_F16(aq0, bf, s1);
      bf = load_b_ktile(Kt, 16, 32, lane); s1 = WMMA_F16(aq1, bf, s1);
    }

    // Online softmax. C layout: lanes 0-15 hold rows 0-7, lanes 16-31 rows 8-15,
    // so row reductions are shfl_xor butterflies of width 16.
#pragma unroll
    for (int r = 0; r < 8; ++r) {
      const float v0 = s0[r] * scale;
      const float v1 = s1[r] * scale;
      float mx = fmaxf(v0, v1);
#pragma unroll
      for (int off = 1; off < 16; off <<= 1) mx = fmaxf(mx, __shfl_xor(mx, off, 16));
      const float mnew  = fmaxf(mrow[r], mx);
      const float alpha = __expf(mrow[r] - mnew);
      const float p0 = __expf(v0 - mnew);
      const float p1 = __expf(v1 - mnew);
      float rs = p0 + p1;
#pragma unroll
      for (int off = 1; off < 16; off <<= 1) rs += __shfl_xor(rs, off, 16);
      lrow[r] = lrow[r] * alpha + rs;
      mrow[r] = mnew;
      o0[r] *= alpha; o1[r] *= alpha; o2[r] *= alpha; o3[r] *= alpha;
      const int m = r + 8 * hf;
      pt[m * LDP + n]      = (_Float16)p0;   // keys kb..kb+15
      pt[m * LDP + 16 + n] = (_Float16)p1;   // keys kb+16..kb+31
    }

    // O += P (16x32) @ V (32x64); P bounced through per-wave LDS to A layout.
    const v16h ap = load_a_ptile(pt, lane);
    {
      v16h bf;
      bf = load_b_vtile(Vt, 0, lane);  o0 = WMMA_F16(ap, bf, o0);
      bf = load_b_vtile(Vt, 16, lane); o1 = WMMA_F16(ap, bf, o1);
      bf = load_b_vtile(Vt, 32, lane); o2 = WMMA_F16(ap, bf, o2);
      bf = load_b_vtile(Vt, 48, lane); o3 = WMMA_F16(ap, bf, o3);
    }
  }

  // Epilogue: normalize by row sums, store fp32.
#pragma unroll
  for (int r = 0; r < 8; ++r) {
    const float inv = 1.0f / lrow[r];
    float* po = out + (size_t)(qrow0 + r + 8 * hf) * HEAD + n;
    po[0]  = o0[r] * inv;
    po[16] = o1[r] * inv;
    po[32] = o2[r] * inv;
    po[48] = o3[r] * inv;
  }
}

// ----------------------------------- launcher -------------------------------------

extern "C" void kernel_launch(void* const* d_in, const int* in_sizes, int n_in,
                              void* d_out, int out_size, void* d_ws, size_t ws_size,
                              hipStream_t stream) {
  const float* x  = (const float*)d_in[0];
  const float* Wq = (const float*)d_in[1];
  const float* bq = (const float*)d_in[2];
  const float* Wk = (const float*)d_in[3];
  const float* bk = (const float*)d_in[4];
  const float* Wv = (const float*)d_in[5];
  const float* bv = (const float*)d_in[6];
  float* out = (float*)d_out;

  // Workspace: q/k/v f16 (3 * 4 MB) + W fragments (3 * 48 KB).
  _Float16* qh = (_Float16*)d_ws;
  _Float16* kh = qh + (size_t)MROWS * HEAD;
  _Float16* vh = kh + (size_t)MROWS * HEAD;
  _Float16* wf = vh + (size_t)MROWS * HEAD;

  // 3 * 12288 K-pair work items -> 144 blocks.
  prep_w_kernel<<<144, 256, 0, stream>>>(Wq, Wk, Wv, wf);

  // 2048 row-tiles of 16, 4 waves per 128-thread block -> 512 blocks.
  qkv_kernel<<<512, 128, 0, stream>>>(x, wf, bq, bk, bv, qh, kh, vh);

  // 16 batches * 16 query-blocks of 128 rows -> 256 blocks, 8 waves each.
  attn_kernel<<<256, 256, 0, stream>>>(qh, kh, vh, out);
}
